// LorentzSparseSqDisAtt_56788057588127
// MI455X (gfx1250) — matrix operand order repaired and checked by reading, not compiled, not run on hardware
//
#include <hip/hip_runtime.h>
#include <hip/hip_bf16.h>

typedef __attribute__((ext_vector_type(16))) __bf16 v16bf;
typedef __attribute__((ext_vector_type(8)))  float  v8f;

// -------------------------------------------------------------------------
// Kernel 1: h-tail matrix g[N][128] where g[i][k] = mx[i][k] for k<127 and
// g[i][127] = t_i = sqrt(sum_k mx[i][k]^2 + 1).  (mx[127] only feeds t.)
// One block = 16 rows of x; 8 waves, wave w computes output cols 16w..16w+15
// via v_wmma_f32_16x16x32_bf16 over K=128 (4 WMMA steps).
// Requires N % 16 == 0 (N = 50000 = 3125 * 16).
// -------------------------------------------------------------------------
__global__ __launch_bounds__(256) void lorentz_linear_wmma(
    const float* __restrict__ x,   // [N,128]
    const float* __restrict__ W,   // [128,128] row-major (out, in)
    const float* __restrict__ bia, // [128]
    float* __restrict__ g,         // [N,128] output
    int N)
{
    __shared__ float lx[16 * 128]; // x tile (fp32)
    __shared__ float lh[16 * 128]; // mx tile (fp32)

    const int tid  = threadIdx.x;
    const int wave = tid >> 5;
    const int lane = tid & 31;
    const long long rowBase = (long long)blockIdx.x * 16;

    // Cooperative load of 16x128 x tile: 512 float4 by 256 threads.
    {
        const float4* src = (const float4*)(x + rowBase * 128);
        float4* dst = (float4*)lx;
        dst[tid]       = src[tid];
        dst[tid + 256] = src[tid + 256];
    }
    __syncthreads();

    const int half = lane >> 4;          // K-half select (wave32 layout)
    const int lrow = lane & 15;          // A row / B column within tile
    const int col  = (wave << 4) + lrow; // global output column 0..127

    v8f acc = {};
    #pragma unroll
    for (int kc = 0; kc < 128; kc += 32) {
        v16bf amat, bmat;
        #pragma unroll
        for (int idx = 0; idx < 16; ++idx) {
            // A 16x32 bf16 layout: VGPR pairs 0..3 -> K {0..7 | 8..15},
            // VGPR pairs 4..7 -> K {16..23 | 24..31} split by lane half.
            const int ka = kc + ((idx >> 3) << 4) + (half << 3) + (idx & 7);
            amat[idx] = (__bf16)lx[lrow * 128 + ka];
            // B 32x16 bf16 layout: same K order as A (K {0..7 | 8..15}, {16..23 | 24..31} by lane half).
            // B[k][n] = W[n][k]  (W is (out,in) row-major).
            const int kb = kc + ((idx >> 3) << 4) + (half << 3) + (idx & 7);
            bmat[idx] = (__bf16)W[col * 128 + kb];
        }
        acc = __builtin_amdgcn_wmma_f32_16x16x32_bf16(
            /*neg_a=*/false, amat, /*neg_b=*/false, bmat,
            /*c_mod=*/(short)0, acc, /*reuse_a=*/false, /*reuse_b=*/false);
    }

    // Bias (per output column) and spill mx tile to LDS.
    // C/D layout: VGPR r holds row M = r + 8*half, col N = lane&15.
    const float bv = bia[col];
    #pragma unroll
    for (int r = 0; r < 8; ++r)
        lh[(r + (half << 3)) * 128 + col] = acc[r] + bv;
    __syncthreads();

    // Row sum-of-squares over all 128 cols: 16 threads per row, 8 cols each,
    // reduced inside each 16-lane half-wave group.
    const int row = tid >> 4;
    const int j   = tid & 15;
    float s = 0.f;
    #pragma unroll
    for (int k2 = 0; k2 < 8; ++k2) {
        const float v = lh[row * 128 + j * 8 + k2];
        s += v * v;
    }
    s += __shfl_xor(s, 1);
    s += __shfl_xor(s, 2);
    s += __shfl_xor(s, 4);
    s += __shfl_xor(s, 8);
    __syncthreads(); // all sumsq reads complete before slot-127 overwrite
    if (j == 0) lh[row * 128 + 127] = sqrtf(s + 1.0f); // 1/c, C = 1
    __syncthreads();

    // Coalesced store of the finished 16x128 g tile.
    {
        float4* dst = (float4*)(g + rowBase * 128);
        const float4* src = (const float4*)lh;
        dst[tid]       = src[tid];
        dst[tid + 256] = src[tid + 256];
    }
}

// -------------------------------------------------------------------------
// Kernel 2: per-edge Lorentz inner product + clipped exp.
// 4 lanes per edge; each lane reads 8 float4 per row (each lane a contiguous
// 128B quarter-row; the 4 lanes are 128B apart per load).  g rows are L2-resident.
// inner = sum_{k<127} gx[k]*gy[k] - gx[127]*gy[127]
// -------------------------------------------------------------------------
__global__ __launch_bounds__(256) void edge_lorentz_att(
    const long long* __restrict__ ei, // [2,E] int64
    const float* __restrict__ g,      // [N,128]
    float* __restrict__ res,          // [E]
    long long E)
{
    const long long tid = (long long)blockIdx.x * 256 + threadIdx.x;
    const long long e   = tid >> 2;
    const int sub       = (int)(tid & 3);
    if (e >= E) return;

    const long long sN = ei[e];
    const long long dN = ei[E + e];
    const float4* gx = (const float4*)(g + sN * 128) + sub * 8;
    const float4* gy = (const float4*)(g + dN * 128) + sub * 8;

    float s = 0.f;
    #pragma unroll
    for (int i = 0; i < 8; ++i) {
        const float4 a = gx[i];
        const float4 b = gy[i];
        s += a.x * b.x + a.y * b.y + a.z * b.z + a.w * b.w;
        if (sub == 3 && i == 7) s -= 2.f * a.w * b.w; // flip sign of the t*t term
    }
    s += __shfl_xor(s, 1);
    s += __shfl_xor(s, 2);

    if (sub == 0) {
        float r = -(1.0f + s);                   // -(C + l_inner)
        r = fminf(fmaxf(r, 1e-10f), 1.0f);       // clip
        res[e] = expf(-r);
    }
}

// -------------------------------------------------------------------------
// Launcher.  Inputs: x [N*128] f32, edge_index [2*E] i64, W [128*128] f32,
// b [128] f32.  Output: edge_index passthrough (2E int64 = 4E float slots)
// followed by res [E] f32.  Scratch d_ws holds g [N*128] f32 (25.6 MB).
// -------------------------------------------------------------------------
extern "C" void kernel_launch(void* const* d_in, const int* in_sizes, int n_in,
                              void* d_out, int out_size, void* d_ws, size_t ws_size,
                              hipStream_t stream) {
    const float*     x  = (const float*)d_in[0];
    const long long* ei = (const long long*)d_in[1];
    const float*     W  = (const float*)d_in[2];
    const float*     b  = (const float*)d_in[3];

    const long long N = (long long)in_sizes[0] / 128; // 50000
    const long long E = (long long)in_sizes[1] / 2;   // 1600000

    float* g   = (float*)d_ws;                        // [N,128]
    float* res = (float*)d_out + 4 * E;               // after 2E int64 (=16E bytes)

    // Passthrough of edge_index into the output buffer (graph-capture safe).
    hipMemcpyAsync(d_out, (const void*)ei, (size_t)(2 * E) * sizeof(long long),
                   hipMemcpyDeviceToDevice, stream);

    // Phase 1: WMMA Lorentz linear (N % 16 == 0 for this problem).
    const int nTiles = (int)(N / 16);
    lorentz_linear_wmma<<<nTiles, 256, 0, stream>>>(x, W, b, g, (int)N);

    // Phase 2: edge attention (4 lanes per edge -> 64 edges per 256-thread block).
    const long long threadsNeeded = 4 * E;
    const int blocks = (int)((threadsNeeded + 255) / 256);
    edge_lorentz_att<<<blocks, 256, 0, stream>>>(ei, g, res, E);
}
